// STGCN_84602265796646
// MI455X (gfx1250) — matrix operand order repaired and checked by reading, not compiled, not run on hardware
//
#include <hip/hip_runtime.h>
#include <hip/hip_bf16.h>

typedef __attribute__((ext_vector_type(16))) _Float16 v16h;
typedef __attribute__((ext_vector_type(8)))  float    v8f;

#define NODE_F 128
#define HID    96
#define TEMP_F 10
#define KPAD   128                      // K for both GEMMs (106 zero-padded for GEMM2)
#define NT     6                        // 96 / 16 column tiles
#define KC     4                        // 128 / 32 K chunks
#define PACKN  (NT * KC * 32 * 16)      // 12288 f16 per packed weight matrix (24 KB)

// ---------------- weight prepack: f32 row-major -> f16 B-fragment order ----------------
// layout: idx = ((nt*KC + kc)*32 + lane)*16 + e
//   k = kc*32 + (lane>>4)*8 + e + (e>=8 ? 8 : 0)   (B frag: lanes 0-15 K{0..7,16..23}, lanes 16-31 K{8..15,24..31})
//   n = nt*16 + (lane&15)
__global__ void k_pack(const float* __restrict__ Wg, const float* __restrict__ W1,
                       _Float16* __restrict__ pg, _Float16* __restrict__ p1) {
  int p = blockIdx.x * blockDim.x + threadIdx.x;
  if (p >= 2 * PACKN) return;
  int mat = p / PACKN;
  int pm  = p - mat * PACKN;
  int e    = pm & 15;
  int lane = (pm >> 4) & 31;
  int kc   = (pm >> 9) & 3;
  int nt   = pm >> 11;
  int k = kc * 32 + (lane >> 4) * 8 + e + (e >= 8 ? 8 : 0);
  int n = nt * 16 + (lane & 15);
  if (mat == 0) {
    pg[pm] = (_Float16)Wg[(size_t)k * HID + n];
  } else {
    p1[pm] = (_Float16)((k < HID + TEMP_F) ? W1[(size_t)k * HID + n] : 0.0f);
  }
}

// ---------------- degree / init ----------------
__global__ void k_init(float* __restrict__ deg, float* __restrict__ g, int n) {
  int i = blockIdx.x * blockDim.x + threadIdx.x;
  if (i < n) deg[i] = 1.0f;            // self-loop contributes 1 to degree
  if (i < n * HID) g[i] = 0.0f;        // scatter accumulator
}

__global__ void k_deg(const int* __restrict__ ei, float* __restrict__ deg, int E) {
  int e = blockIdx.x * blockDim.x + threadIdx.x;
  if (e < E) atomicAdd(&deg[ei[E + e]], 1.0f);   // dst = edge_index[1][e]
}

__global__ void k_dinv(float* __restrict__ deg, int n) {
  int i = blockIdx.x * blockDim.x + threadIdx.x;
  if (i < n) deg[i] = rsqrtf(deg[i]);            // deg >= 1 always
}

// helper: stage 24 KB of packed weights into LDS with b128 copies
__device__ __forceinline__ void stage_weights(const _Float16* __restrict__ pack,
                                              _Float16* __restrict__ lds) {
  const uint4* s = (const uint4*)pack;
  uint4* d = (uint4*)lds;
  for (int i = threadIdx.x; i < PACKN / 8; i += blockDim.x) d[i] = s[i];
  __syncthreads();
}

// ---------------- GEMM 1: h = x @ W_gcn (f16 WMMA, f32 accum) ----------------
__global__ void k_gemm_gcn(const float* __restrict__ x, const _Float16* __restrict__ Wp,
                           float* __restrict__ h, int n) {
  __shared__ __align__(16) _Float16 lds[PACKN];
  stage_weights(Wp, lds);

  const int wave  = blockIdx.x * (blockDim.x >> 5) + (threadIdx.x >> 5);
  const int tiles = (n + 15) >> 4;
  if (wave >= tiles) return;                 // wave-uniform: EXEC stays all-ones
  const int lane = threadIdx.x & 31;
  const int row0 = wave << 4;
  const int l15  = lane & 15;
  const int hi   = lane >> 4;
  const int kbA  = hi * 8;                   // A frag: lanes 0-15 K{0..7,16..23}, hi +8

  int arow = row0 + l15; if (arow >= n) arow = n - 1;
  const float* xrow = x + (size_t)arow * NODE_F;

  v8f acc[NT] = {};
  for (int kc = 0; kc < KC; ++kc) {
    const int k0 = kc * 32;
    const float4* xr4 = (const float4*)(xrow + k0 + kbA);
    float4 p0 = xr4[0], p1 = xr4[1];         // K+0..7
    float4 q0 = xr4[4], q1 = xr4[5];         // K+16..23
    float af[16] = {p0.x, p0.y, p0.z, p0.w, p1.x, p1.y, p1.z, p1.w,
                    q0.x, q0.y, q0.z, q0.w, q1.x, q1.y, q1.z, q1.w};
    v16h a;
#pragma unroll
    for (int e = 0; e < 16; ++e) a[e] = (_Float16)af[e];
#pragma unroll
    for (int nt = 0; nt < NT; ++nt) {
      v16h b = *(const v16h*)&lds[(size_t)((nt * KC + kc) * 32 + lane) * 16];
      acc[nt] = __builtin_amdgcn_wmma_f32_16x16x32_f16(false, a, false, b,
                                                       (short)0, acc[nt], false, false);
    }
  }
  const int mb = hi * 8;                     // C/D: lanes 0-15 rows 0-7, hi rows 8-15
#pragma unroll
  for (int nt = 0; nt < NT; ++nt) {
#pragma unroll
    for (int v = 0; v < 8; ++v) {
      int r = row0 + mb + v;
      if (r < n) h[(size_t)r * HID + nt * 16 + l15] = acc[nt][v];
    }
  }
}

// ---------------- edge scatter: g[dst] += h[src] * dinv[src]*dinv[dst] ----------------
__global__ void k_scatter(const int* __restrict__ ei, const float* __restrict__ h,
                          const float* __restrict__ dinv, float* __restrict__ g, int E) {
  const int wave = blockIdx.x * (blockDim.x >> 5) + (threadIdx.x >> 5);
  if (wave >= E) return;
  const int lane = threadIdx.x & 31;
  const int src = ei[wave];
  const int dst = ei[E + wave];
  const float norm = dinv[src] * dinv[dst];
  const float* hs = h + (size_t)src * HID;
  float* gd = g + (size_t)dst * HID;
#pragma unroll
  for (int c = 0; c < 3; ++c) {              // 96 = 3 * 32 lanes, fully coalesced
    int f = lane + 32 * c;
    atomicAdd(&gd[f], hs[f] * norm);
  }
}

// ---------------- GCN epilogue: g = relu(g + h*dinv^2 + b_gcn) ----------------
__global__ void k_gcn_fin(float* __restrict__ g, const float* __restrict__ h,
                          const float* __restrict__ dinv, const float* __restrict__ bg, int n) {
  int i = blockIdx.x * blockDim.x + threadIdx.x;
  if (i >= n * HID) return;
  int node = i / HID, f = i - node * HID;
  float d = dinv[node];
  g[i] = fmaxf(g[i] + h[i] * d * d + bg[f], 0.0f);
}

// ---------------- GEMM 2: t = relu([g | temporal | 0pad] @ W1 + b1) ----------------
__global__ void k_gemm_mlp(const float* __restrict__ g, const float* __restrict__ temporal,
                           const _Float16* __restrict__ Wp, const float* __restrict__ b1,
                           float* __restrict__ t, int n) {
  __shared__ __align__(16) _Float16 lds[PACKN];
  stage_weights(Wp, lds);

  const int wave  = blockIdx.x * (blockDim.x >> 5) + (threadIdx.x >> 5);
  const int tiles = (n + 15) >> 4;
  if (wave >= tiles) return;
  const int lane = threadIdx.x & 31;
  const int row0 = wave << 4;
  const int l15  = lane & 15;
  const int hi   = lane >> 4;
  const int kbA  = hi * 8;
  int arow = row0 + l15; if (arow >= n) arow = n - 1;
  const float* gr = g + (size_t)arow * HID;
  const float* tr = temporal + (size_t)arow * TEMP_F;

  v8f acc[NT] = {};
  for (int kc = 0; kc < KC; ++kc) {
    const int k0 = kc * 32;
    v16h a;
#pragma unroll
    for (int e = 0; e < 16; ++e) {
      int k = k0 + kbA + (e < 8 ? e : e + 8);
      float av = 0.0f;
      if (k < HID)               av = gr[k];
      else if (k < HID + TEMP_F) av = tr[k - HID];
      a[e] = (_Float16)av;
    }
#pragma unroll
    for (int nt = 0; nt < NT; ++nt) {
      v16h b = *(const v16h*)&lds[(size_t)((nt * KC + kc) * 32 + lane) * 16];
      acc[nt] = __builtin_amdgcn_wmma_f32_16x16x32_f16(false, a, false, b,
                                                       (short)0, acc[nt], false, false);
    }
  }
  const int mb = hi * 8;
#pragma unroll
  for (int nt = 0; nt < NT; ++nt) {
    float bias = b1[nt * 16 + l15];
#pragma unroll
    for (int v = 0; v < 8; ++v) {
      int r = row0 + mb + v;
      if (r < n) t[(size_t)r * HID + nt * 16 + l15] = fmaxf(acc[nt][v] + bias, 0.0f);
    }
  }
}

// ---------------- final: out = relu([t | x] . W2 + b2), one wave per node ----------------
__global__ void k_final(const float* __restrict__ t, const float* __restrict__ x,
                        const float* __restrict__ W2, const float* __restrict__ b2,
                        float* __restrict__ out, int n) {
  const int wave = blockIdx.x * (blockDim.x >> 5) + (threadIdx.x >> 5);
  if (wave >= n) return;
  const int lane = threadIdx.x & 31;
  float s = 0.0f;
#pragma unroll
  for (int c = 0; c < 7; ++c) {              // 224 = 7 * 32
    int j = lane + 32 * c;
    float a = (j < HID) ? t[(size_t)wave * HID + j]
                        : x[(size_t)wave * NODE_F + (j - HID)];
    s += a * W2[j];
  }
#pragma unroll
  for (int off = 16; off > 0; off >>= 1) s += __shfl_xor(s, off, 32);
  if (lane == 0) out[wave] = fmaxf(s + b2[0], 0.0f);
}

extern "C" void kernel_launch(void* const* d_in, const int* in_sizes, int n_in,
                              void* d_out, int out_size, void* d_ws, size_t ws_size,
                              hipStream_t stream) {
  const float* x        = (const float*)d_in[0];
  const int*   ei       = (const int*)  d_in[1];
  const float* temporal = (const float*)d_in[2];
  const float* Wg       = (const float*)d_in[3];
  const float* bg       = (const float*)d_in[4];
  const float* W1       = (const float*)d_in[5];
  const float* b1       = (const float*)d_in[6];
  const float* W2       = (const float*)d_in[7];
  const float* b2       = (const float*)d_in[8];
  float* out = (float*)d_out;

  const int Nn = in_sizes[0] / NODE_F;   // 50000
  const int Ee = in_sizes[1] / 2;        // 800000

  // workspace layout (packs first for 16B alignment)
  _Float16* pg = (_Float16*)d_ws;                 // PACKN f16 (24 KB)
  _Float16* p1 = pg + PACKN;                      // PACKN f16 (24 KB)
  float* deg = (float*)(p1 + PACKN);              // N floats (becomes dinv in-place)
  float* h   = deg + Nn;                          // N*HID
  float* g   = h + (size_t)Nn * HID;              // N*HID
  float* t   = h;                                 // reuse h's region after GCN is done

  const int B   = 256;                            // 8 wave32 per block
  const int wpb = B / 32;
  const int tiles = (Nn + 15) / 16;

  k_pack    <<<(2 * PACKN + B - 1) / B, B, 0, stream>>>(Wg, W1, pg, p1);
  k_init    <<<(Nn * HID + B - 1) / B,  B, 0, stream>>>(deg, g, Nn);
  k_deg     <<<(Ee + B - 1) / B,        B, 0, stream>>>(ei, deg, Ee);
  k_dinv    <<<(Nn + B - 1) / B,        B, 0, stream>>>(deg, Nn);
  k_gemm_gcn<<<(tiles + wpb - 1) / wpb, B, 0, stream>>>(x, pg, h, Nn);
  k_scatter <<<(Ee + wpb - 1) / wpb,    B, 0, stream>>>(ei, h, deg, g, Ee);
  k_gcn_fin <<<(Nn * HID + B - 1) / B,  B, 0, stream>>>(g, h, deg, bg, Nn);
  k_gemm_mlp<<<(tiles + wpb - 1) / wpb, B, 0, stream>>>(g, temporal, p1, b1, t, Nn);
  k_final   <<<(Nn + wpb - 1) / wpb,    B, 0, stream>>>(t, x, W2, b2, out, Nn);
}
